// Attention_9740985828113
// MI455X (gfx1250) — compile-verified, matrix operand order checked by reading, not run on hardware
//
#include <hip/hip_runtime.h>
#include <hip/hip_bf16.h>

#ifndef __has_builtin
#define __has_builtin(x) 0
#endif

typedef __attribute__((ext_vector_type(16))) __bf16 v16bf;
typedef __attribute__((ext_vector_type(8)))  __bf16 v8bf;
typedef __attribute__((ext_vector_type(8)))  float  v8f;
typedef __attribute__((ext_vector_type(4)))  int    v4i;

#define BIG_NEG (-2.3819763e+38f)

// ---------------------------------------------------------------------------
// helpers
// ---------------------------------------------------------------------------
__device__ __forceinline__ v16bf load_frag(const __bf16* p0, const __bf16* p1) {
  v8bf lo = *(const v8bf*)p0;
  v8bf hi = *(const v8bf*)p1;
  return __builtin_shufflevector(lo, hi, 0,1,2,3,4,5,6,7,8,9,10,11,12,13,14,15);
}

__device__ __forceinline__ v8f wmma_bf16(v16bf a, v16bf b, v8f c) {
  return __builtin_amdgcn_wmma_f32_16x16x32_bf16(false, a, false, b, (short)0, c,
                                                 false, false);
}

#if __has_builtin(__builtin_amdgcn_global_load_async_to_lds_b128)
#define HAVE_ASYNC_LDS 1
#else
#define HAVE_ASYNC_LDS 0
#endif

typedef __attribute__((address_space(1))) v4i GV4;
typedef __attribute__((address_space(3))) v4i LV4;

__device__ __forceinline__ void async_copy_b128(const void* g, void* l) {
#if HAVE_ASYNC_LDS
  void* gnc = const_cast<void*>(g);
  __builtin_amdgcn_global_load_async_to_lds_b128((GV4*)gnc, (LV4*)l, 0, 0);
#else
  *(v8bf*)l = *(const v8bf*)g;
#endif
}

__device__ __forceinline__ void wait_async_all() {
#if HAVE_ASYNC_LDS
#if __has_builtin(__builtin_amdgcn_s_wait_asynccnt)
  __builtin_amdgcn_s_wait_asynccnt(0);
#else
  asm volatile("s_wait_asynccnt 0" ::: "memory");
#endif
#endif
}

__device__ __forceinline__ void wait_lds_writes() {
#if __has_builtin(__builtin_amdgcn_s_wait_dscnt)
  __builtin_amdgcn_s_wait_dscnt(0);
#else
  asm volatile("s_wait_dscnt 0" ::: "memory");
#endif
}

// ---------------------------------------------------------------------------
// cast / pack kernels (f32 -> bf16, weight transposes for WMMA B-frag layout)
// ---------------------------------------------------------------------------
__global__ __launch_bounds__(256) void cast_bf16_kernel(const float* __restrict__ in,
                                                        __bf16* __restrict__ out,
                                                        int n) {
  int i = blockIdx.x * 256 + threadIdx.x;
  if (i < n) out[i] = (__bf16)in[i];
}

// q_w (N,D,H) -> qwb[(n*H+h)][d]   (row-major, rows = output col, cols = k)
__global__ __launch_bounds__(256) void pack_qw_kernel(const float* __restrict__ qw,
                                                      __bf16* __restrict__ qwb) {
  int i = blockIdx.x * 256 + threadIdx.x;  // i < 8*2048*256
  int h = i & 255, d = (i >> 8) & 2047, n = i >> 19;
  qwb[((size_t)(n * 256 + h)) * 2048 + d] = (__bf16)qw[i];
}

// kv_w (2,1,D,H) -> kvwb[(c*H+h)][d]
__global__ __launch_bounds__(256) void pack_kvw_kernel(const float* __restrict__ kvw,
                                                       __bf16* __restrict__ kvwb) {
  int i = blockIdx.x * 256 + threadIdx.x;  // i < 2*2048*256
  int h = i & 255, d = (i >> 8) & 2047, c = i >> 19;
  kvwb[((size_t)(c * 256 + h)) * 2048 + d] = (__bf16)kvw[i];
}

// out_w (N,H,D) -> owbt[d][(n*H+h)]
__global__ __launch_bounds__(256) void pack_ow_kernel(const float* __restrict__ ow,
                                                      __bf16* __restrict__ owbt) {
  int i = blockIdx.x * 256 + threadIdx.x;  // i < 2048*2048
  int d = i & 2047, nh = i >> 11;
  owbt[(size_t)d * 2048 + nh] = (__bf16)ow[i];
}

// ---------------------------------------------------------------------------
// Generic bf16 GEMM: C(MxN) = A(MxK) * Bt(NxK)^T, LDS tiled 128x128x32.
// 8 waves per block; wave tile 32x64 = 2x4 WMMA 16x16 tiles.
// ---------------------------------------------------------------------------
template <bool OUT_BF16>
__global__ __launch_bounds__(256) void gemm_bf16_kernel(
    const __bf16* __restrict__ A, const __bf16* __restrict__ Bt,
    void* __restrict__ C, int M, int N, int Kd) {
  __shared__ __bf16 As[128 * 32];
  __shared__ __bf16 Bs[128 * 32];

  const int tid = threadIdx.x;
  const int lane = tid & 31, w = tid >> 5;
  const int wm = w & 3, wn = w >> 2;  // wave grid 4x2: rows wm*32, cols wn*64
  const int hlf = lane >> 4, l16 = lane & 15;
  const int block_m = blockIdx.y * 128, block_n = blockIdx.x * 128;

  v8f acc[2][4] = {};
  for (int k0 = 0; k0 < Kd; k0 += 32) {
    // stage tiles: 512 chunks of 8 bf16 each, 2 per thread per matrix
    for (int i = 0; i < 2; ++i) {
      int c = tid + i * 256;
      int r = c >> 2, cc = (c & 3) * 8;
      *(v8bf*)&As[r * 32 + cc] =
          *(const v8bf*)&A[(size_t)(block_m + r) * Kd + k0 + cc];
      *(v8bf*)&Bs[r * 32 + cc] =
          *(const v8bf*)&Bt[(size_t)(block_n + r) * Kd + k0 + cc];
    }
    __syncthreads();

    v16bf af[2], bfg[4];
    for (int mi = 0; mi < 2; ++mi) {
      int m = wm * 32 + mi * 16 + l16;
      af[mi] = load_frag(&As[m * 32 + 8 * hlf], &As[m * 32 + 16 + 8 * hlf]);
    }
    for (int ni = 0; ni < 4; ++ni) {
      int n = wn * 64 + ni * 16 + l16;
      bfg[ni] = load_frag(&Bs[n * 32 + 8 * hlf], &Bs[n * 32 + 16 + 8 * hlf]);
    }
    for (int mi = 0; mi < 2; ++mi)
      for (int ni = 0; ni < 4; ++ni)
        acc[mi][ni] = wmma_bf16(af[mi], bfg[ni], acc[mi][ni]);
    __syncthreads();
  }

  for (int mi = 0; mi < 2; ++mi)
    for (int ni = 0; ni < 4; ++ni)
      for (int r = 0; r < 8; ++r) {
        int mg = block_m + wm * 32 + mi * 16 + r + 8 * hlf;
        int ng = block_n + wn * 64 + ni * 16 + l16;
        float v = acc[mi][ni][r];
        if (OUT_BF16)
          ((__bf16*)C)[(size_t)mg * N + ng] = (__bf16)v;
        else
          ((float*)C)[(size_t)mg * N + ng] = v;
      }
}

// ---------------------------------------------------------------------------
// RoPE (in-place on bf16): Q gets H^-0.5 scale folded in, K does not.
// ---------------------------------------------------------------------------
__global__ __launch_bounds__(256) void rope_q_kernel(__bf16* __restrict__ Qb,
                                                     const int* __restrict__ pos) {
  int i = blockIdx.x * 256 + threadIdx.x;  // i < 2*2048*8*128 = 2^22
  int j = i & 127, n = (i >> 7) & 7, t = (i >> 10) & 2047, b = i >> 21;
  float p = (float)pos[b * 2048 + t];
  float fe = (2.0f / 256.0f) * (float)j;
  float rad = p / __powf(10000.0f, fe);
  float s, c;
  __sincosf(rad, &s, &c);
  size_t base = ((size_t)(b * 2048 + t)) * 2048 + n * 256;
  float x1 = (float)Qb[base + j], x2 = (float)Qb[base + 128 + j];
  const float scale = 0.0625f;  // 256^-0.5
  Qb[base + j] = (__bf16)((x1 * c - x2 * s) * scale);
  Qb[base + 128 + j] = (__bf16)((x2 * c + x1 * s) * scale);
}

__global__ __launch_bounds__(256) void rope_k_kernel(__bf16* __restrict__ KVb,
                                                     const int* __restrict__ pos) {
  int i = blockIdx.x * 256 + threadIdx.x;  // i < 2*2048*128 = 2^19
  int j = i & 127, t = (i >> 7) & 2047, b = i >> 18;
  float p = (float)pos[b * 2048 + t];
  float fe = (2.0f / 256.0f) * (float)j;
  float rad = p / __powf(10000.0f, fe);
  float s, c;
  __sincosf(rad, &s, &c);
  size_t base = ((size_t)(b * 2048 + t)) * 512;  // K occupies cols [0,256)
  float x1 = (float)KVb[base + j], x2 = (float)KVb[base + 128 + j];
  KVb[base + j] = (__bf16)(x1 * c - x2 * s);
  KVb[base + 128 + j] = (__bf16)(x2 * c + x1 * s);
}

// ---------------------------------------------------------------------------
// Flash attention: grid (T/128, B*N). 8 waves/block, each wave owns 16 q rows.
// Q fragments re-streamed from L2 per chunk (keeps VGPRs < 256, no MSB churn);
// K/V async-staged to LDS per 32-key chunk. Online softmax with tanh softcap;
// P transposed via LDS into A-frag layout.
// ---------------------------------------------------------------------------
__global__ __launch_bounds__(256) void attn_kernel(const __bf16* __restrict__ Qb,
                                                   const __bf16* __restrict__ KVb,
                                                   __bf16* __restrict__ Enc) {
  __shared__ __bf16 Ks[32 * 256];  // 16 KB
  __shared__ __bf16 Vs[32 * 256];  // 16 KB
  __shared__ __bf16 Vt[256 * 32];  // 16 KB (V transposed)
  __shared__ __bf16 Ps[8][16 * 32];  // 8 KB (per-wave P tile)

  const int tid = threadIdx.x, lane = tid & 31, w = tid >> 5;
  const int hlf = lane >> 4, l16 = lane & 15;
  const int hb = blockIdx.y;  // 0..15
  const int b = hb >> 3, n = hb & 7;
  const int m0 = blockIdx.x * 128 + w * 16;  // wave's first query row (t)

  // per-lane Q row base (row = t for this lane's A-frag slot)
  const size_t qbase = ((size_t)(b * 2048) + m0 + l16) * 2048 + n * 256;

  v8f o[16] = {};
  float mrow[8], lrow[8];
  for (int r = 0; r < 8; ++r) {
    mrow[r] = -__builtin_inff();
    lrow[r] = 0.0f;
  }

  const int nchunks = blockIdx.x * 4 + 4;  // keys up to block's last row
  for (int sc = 0; sc < nchunks; ++sc) {
    const int s0 = sc * 32;
    // --- stage K and V (32 rows x 256 bf16 each) via async b128 copies ---
    for (int i = 0; i < 4; ++i) {
      int c = tid * 4 + i;  // 0..1023, 8 bf16 per chunk
      int sr = c >> 5, sco = (c & 31) * 8;
      size_t g = ((size_t)(b * 2048) + s0 + sr) * 512;
      async_copy_b128(KVb + g + sco, &Ks[sr * 256 + sco]);
      async_copy_b128(KVb + g + 256 + sco, &Vs[sr * 256 + sco]);
    }
    wait_async_all();
    __syncthreads();
    // --- transpose V into Vt (for B-frag contiguous reads) ---
    for (int i = tid; i < 32 * 256; i += 256) {
      int s = i >> 8, h = i & 255;
      Vt[h * 32 + s] = Vs[i];
    }
    __syncthreads();

    const bool active = (s0 <= m0);
    if (active) {
      // --- logits: Q(16x256) * K^T(256x32); Q frags streamed from L2 ---
      v8f cl0 = {}, cl1 = {};
      for (int hs = 0; hs < 8; ++hs) {
        const __bf16* q = Qb + qbase + hs * 32;
        v16bf qfrag = load_frag(q + 8 * hlf, q + 16 + 8 * hlf);
        const __bf16* k0p = &Ks[l16 * 256 + hs * 32];
        const __bf16* k1p = &Ks[(16 + l16) * 256 + hs * 32];
        v16bf kb0 = load_frag(k0p + 8 * hlf, k0p + 16 + 8 * hlf);
        v16bf kb1 = load_frag(k1p + 8 * hlf, k1p + 16 + 8 * hlf);
        cl0 = wmma_bf16(qfrag, kb0, cl0);
        cl1 = wmma_bf16(qfrag, kb1, cl1);
      }
      float lg[2][8];
      for (int r = 0; r < 8; ++r) {
        int tq = m0 + r + 8 * hlf;
        float L0 = tanhf(cl0[r] * 0.02f) * 50.0f;  // softcap
        float L1 = tanhf(cl1[r] * 0.02f) * 50.0f;
        lg[0][r] = (s0 + l16 <= tq) ? L0 : BIG_NEG;
        lg[1][r] = (s0 + 16 + l16 <= tq) ? L1 : BIG_NEG;
      }
      // --- online softmax update ---
      float a8[8];
      for (int r = 0; r < 8; ++r) {
        float vm = fmaxf(lg[0][r], lg[1][r]);
        for (int msk = 1; msk < 16; msk <<= 1)
          vm = fmaxf(vm, __shfl_xor(vm, msk, 32));
        float mn = fmaxf(mrow[r], vm);
        float alpha = __expf(mrow[r] - mn);
        float p0 = __expf(lg[0][r] - mn);
        float p1 = __expf(lg[1][r] - mn);
        float ps = p0 + p1;
        for (int msk = 1; msk < 16; msk <<= 1) ps += __shfl_xor(ps, msk, 32);
        lrow[r] = lrow[r] * alpha + ps;
        mrow[r] = mn;
        a8[r] = alpha;
        lg[0][r] = p0;  // reuse as probabilities
        lg[1][r] = p1;
      }
      for (int ot = 0; ot < 16; ++ot)
        for (int r = 0; r < 8; ++r) o[ot][r] *= a8[r];

      // --- transpose P (16x32) through LDS into A-frag layout ---
      for (int st = 0; st < 2; ++st)
        for (int r = 0; r < 8; ++r)
          Ps[w][(r + 8 * hlf) * 32 + st * 16 + l16] = (__bf16)lg[st][r];
      wait_lds_writes();  // same-wave DS ordering; private per-wave tile
      v16bf pa = load_frag(&Ps[w][l16 * 32 + 8 * hlf],
                           &Ps[w][l16 * 32 + 16 + 8 * hlf]);

      // --- O(16x256) += P(16x32) * V(32x256) ---
      for (int ot = 0; ot < 16; ++ot) {
        int h = ot * 16 + l16;
        v16bf vb = load_frag(&Vt[h * 32 + 8 * hlf], &Vt[h * 32 + 16 + 8 * hlf]);
        o[ot] = wmma_bf16(pa, vb, o[ot]);
      }
    }
    __syncthreads();  // protect Ks/Vs/Vt before next chunk restages
  }

  // --- epilogue: normalize and store encoded (B,T,N,H) as bf16 ---
  float inv[8];
  for (int r = 0; r < 8; ++r) inv[r] = 1.0f / lrow[r];
  for (int ot = 0; ot < 16; ++ot)
    for (int r = 0; r < 8; ++r) {
      int tq = m0 + r + 8 * hlf;
      int h = ot * 16 + l16;
      Enc[((size_t)(b * 2048 + tq)) * 2048 + n * 256 + h] =
          (__bf16)(o[ot][r] * inv[r]);
    }
}

// ---------------------------------------------------------------------------
// host launcher
// ---------------------------------------------------------------------------
extern "C" void kernel_launch(void* const* d_in, const int* in_sizes, int n_in,
                              void* d_out, int out_size, void* d_ws,
                              size_t ws_size, hipStream_t stream) {
  (void)in_sizes; (void)n_in; (void)out_size; (void)ws_size;
  const float* x = (const float*)d_in[0];
  const int* positions = (const int*)d_in[1];
  // d_in[2] = attn_mask (causal tril) — implemented analytically
  const float* q_w = (const float*)d_in[3];
  const float* kv_w = (const float*)d_in[4];
  const float* out_w = (const float*)d_in[5];

  char* ws = (char*)d_ws;
  size_t off = 0;
  auto take = [&](size_t elems) {
    void* p = ws + off;
    off = (off + elems * 2 + 255) & ~(size_t)255;
    return p;
  };
  __bf16* xb   = (__bf16*)take((size_t)4096 * 2048);  // x as bf16
  __bf16* qwb  = (__bf16*)take((size_t)2048 * 2048);  // q_w packed (nh, d)
  __bf16* kvwb = (__bf16*)take((size_t)512 * 2048);   // kv_w packed (c*H+h, d)
  __bf16* owbt = (__bf16*)take((size_t)2048 * 2048);  // out_w packed (d, nh)
  __bf16* Qb   = (__bf16*)take((size_t)4096 * 2048);  // Q (B,T,N,H)
  __bf16* KVb  = (__bf16*)take((size_t)4096 * 512);   // [K | V] per row
  __bf16* Enc  = (__bf16*)take((size_t)4096 * 2048);  // encoded (B,T,N,H)

  cast_bf16_kernel<<<8388608 / 256, 256, 0, stream>>>(x, xb, 8388608);
  pack_qw_kernel<<<4194304 / 256, 256, 0, stream>>>(q_w, qwb);
  pack_kvw_kernel<<<1048576 / 256, 256, 0, stream>>>(kv_w, kvwb);
  pack_ow_kernel<<<4194304 / 256, 256, 0, stream>>>(out_w, owbt);

  gemm_bf16_kernel<true><<<dim3(16, 32), 256, 0, stream>>>(xb, qwb, Qb, 4096,
                                                           2048, 2048);
  gemm_bf16_kernel<true><<<dim3(4, 32), 256, 0, stream>>>(xb, kvwb, KVb, 4096,
                                                          512, 2048);
  rope_q_kernel<<<16384, 256, 0, stream>>>(Qb, positions);
  rope_k_kernel<<<2048, 256, 0, stream>>>(KVb, positions);

  attn_kernel<<<dim3(16, 16), 256, 0, stream>>>(Qb, KVb, Enc);

  gemm_bf16_kernel<false><<<dim3(16, 32), 256, 0, stream>>>(Enc, owbt, d_out,
                                                            4096, 2048, 2048);
}
